// KMeansEMA_5592047419507
// MI455X (gfx1250) — matrix operand order repaired and checked by reading, not compiled, not run on hardware
//
#include <hip/hip_runtime.h>
#include <math.h>

// ---------------------------------------------------------------------------
// VQ-VAE codebook (k-means EMA) for MI455X / gfx1250.
// Dominant cost: 65536x1024x256 fp32 GEMM for nearest-codebook search.
// Strategy: split-f16 (hi/lo) emulated-fp32 GEMM on v_wmma_f32_16x16x32_f16,
// fused with the argmax; everything else is bandwidth-trivial.
// ---------------------------------------------------------------------------

typedef _Float16 v16h __attribute__((ext_vector_type(16)));
typedef _Float16 v8h  __attribute__((ext_vector_type(8)));
typedef float    v8f  __attribute__((ext_vector_type(8)));
typedef float    v4f  __attribute__((ext_vector_type(4)));

#define K_CODES   1024
#define D_DIM     256
#define N_ROWS    65536          // 16 * 4096
#define DECAY_F   0.8f
#define OMD_F     0.2f
#define EPS_F     1e-5f
#define EROW      264            // padded LDS row stride (halves): bank-conflict free

// -------------------------------- zero scratch -----------------------------
__global__ void vq_zero(float* __restrict__ p, int n) {
  int i = blockIdx.x * blockDim.x + threadIdx.x;
  if (i < n) p[i] = 0.0f;
}

// ------------------------- 0.5 * |embed_k|^2 -------------------------------
__global__ void vq_norms(const float* __restrict__ embed, float* __restrict__ norms05) {
  int k = blockIdx.x * blockDim.x + threadIdx.x;
  if (k >= K_CODES) return;
  const v4f* e = (const v4f*)(embed + (size_t)k * D_DIM);
  float s = 0.0f;
#pragma unroll 8
  for (int i = 0; i < D_DIM / 4; ++i) {
    v4f v = e[i];
    s += v[0]*v[0] + v[1]*v[1] + v[2]*v[2] + v[3]*v[3];
  }
  norms05[k] = 0.5f * s;
}

// --------------- fused distance GEMM (split-f16 WMMA) + argmax -------------
// Block: 256 threads = 8 waves; each wave owns 16 rows (block tile = 128 rows).
// A (x tile, 16xD) register-resident as hi/lo f16 fragments for the whole K
// sweep; B (16 codes x D of embed) staged per-block in LDS, pre-split hi/lo.
__global__ __launch_bounds__(256, 1) void vq_argmax(
    const float* __restrict__ x, const float* __restrict__ embed,
    const float* __restrict__ norms05, int* __restrict__ ind)
{
  __shared__ __align__(16) _Float16 ehi[16 * EROW];
  __shared__ __align__(16) _Float16 elo[16 * EROW];

  const int tid   = threadIdx.x;
  const int lane  = tid & 31;
  const int wave  = tid >> 5;
  const int m     = lane & 15;     // row-within-tile (A) / column n (B,C)
  const int khalf = lane >> 4;     // K-dim half selector per WMMA layout
  const int rowbase = blockIdx.x * 128 + wave * 16;
  const int row     = rowbase + m;

  // ---- load A tile once, split fp32 -> f16 hi + f16 lo, WMMA A layout ----
  // A 16x32 f16 layout: lane m, VGPR e: e<8 -> k = khalf*8 + e ;
  //                                e>=8 -> k = 16 + khalf*8 + (e-8)
  v16h ahi[8], alo[8];
  {
    const float* xr = x + (size_t)row * D_DIM;
#pragma unroll
    for (int c = 0; c < 8; ++c) {
      const int d0 = c * 32 + khalf * 8;
      v4f f0 = *(const v4f*)(xr + d0);
      v4f f1 = *(const v4f*)(xr + d0 + 4);
      v4f g0 = *(const v4f*)(xr + d0 + 16);
      v4f g1 = *(const v4f*)(xr + d0 + 20);
      float t[16];
#pragma unroll
      for (int j = 0; j < 4; ++j) { t[j] = f0[j]; t[4+j] = f1[j]; t[8+j] = g0[j]; t[12+j] = g1[j]; }
#pragma unroll
      for (int e = 0; e < 16; ++e) {
        _Float16 h = (_Float16)t[e];
        ahi[c][e] = h;
        alo[c][e] = (_Float16)(t[e] - (float)h);
      }
    }
  }

  float bestv[8];
  int   besti[8];
#pragma unroll
  for (int r = 0; r < 8; ++r) { bestv[r] = -3.4e38f; besti[r] = 0; }

  for (int kc = 0; kc < K_CODES / 16; ++kc) {
    __syncthreads();  // previous chunk's LDS fully consumed
    // ---- stage 16 embed rows, pre-split hi/lo (coalesced: 16 thr/row) ----
    {
      const int r  = tid >> 4;          // 0..15 : code row within chunk
      const int dg = (tid & 15) * 16;   // 0..240
      const float* er = embed + ((size_t)(kc * 16 + r)) * D_DIM + dg;
#pragma unroll
      for (int q = 0; q < 4; ++q) {
        v4f v = *(const v4f*)(er + q * 4);
#pragma unroll
        for (int j = 0; j < 4; ++j) {
          float f = v[j];
          _Float16 h = (_Float16)f;
          ehi[r * EROW + dg + q * 4 + j] = h;
          elo[r * EROW + dg + q * 4 + j] = (_Float16)(f - (float)h);
        }
      }
    }
    if (kc + 1 < K_CODES / 16)   // gfx1250 global_prefetch of next chunk
      __builtin_prefetch(embed + ((size_t)(kc + 1) * 16) * D_DIM + (size_t)(tid & 15) * 64, 0, 1);
    __syncthreads();

    // ---- split-fp32 dot: hi*hi + hi*lo + lo*hi, f32 accumulate ----------
    v8f acc = {0.f, 0.f, 0.f, 0.f, 0.f, 0.f, 0.f, 0.f};
#pragma unroll
    for (int c = 0; c < 8; ++c) {
      // B 32x16 f16 layout: lane col n=m, e<8 -> k = khalf*8 + e ; e>=8 -> k = 16 + khalf*8 + (e-8)
      const int d0 = c * 32 + khalf * 8;
      const _Float16* ph = ehi + m * EROW + d0;
      const _Float16* pl = elo + m * EROW + d0;
      v8h h0 = *(const v8h*)(ph);
      v8h h1 = *(const v8h*)(ph + 16);
      v8h l0 = *(const v8h*)(pl);
      v8h l1 = *(const v8h*)(pl + 16);
      v16h bh = __builtin_shufflevector(h0, h1, 0,1,2,3,4,5,6,7,8,9,10,11,12,13,14,15);
      v16h bl = __builtin_shufflevector(l0, l1, 0,1,2,3,4,5,6,7,8,9,10,11,12,13,14,15);
      acc = __builtin_amdgcn_wmma_f32_16x16x32_f16(false, ahi[c], false, bh, (short)0, acc, false, false);
      acc = __builtin_amdgcn_wmma_f32_16x16x32_f16(false, ahi[c], false, bl, (short)0, acc, false, false);
      acc = __builtin_amdgcn_wmma_f32_16x16x32_f16(false, alo[c], false, bh, (short)0, acc, false, false);
    }

    // C layout: VGPR r -> row (rowbase + khalf*8 + r), lane%16 -> code col
    const int   code = kc * 16 + m;
    const float nrm  = norms05[code];
#pragma unroll
    for (int r = 0; r < 8; ++r) {
      float v = acc[r] - nrm;             // argmax of (x.e - 0.5|e|^2)
      if (v > bestv[r]) { bestv[r] = v; besti[r] = code; }  // strict > = first-index ties
    }
  }

  // ---- argmax reduction across the 16 lanes of each half (xor<16) --------
#pragma unroll
  for (int off = 8; off >= 1; off >>= 1) {
#pragma unroll
    for (int r = 0; r < 8; ++r) {
      float ov = __shfl_xor(bestv[r], off, 32);
      int   oi = __shfl_xor(besti[r], off, 32);
      if (ov > bestv[r] || (ov == bestv[r] && oi < besti[r])) { bestv[r] = ov; besti[r] = oi; }
    }
  }
  if (m == 0) {
    const int base = rowbase + khalf * 8;
#pragma unroll
    for (int r = 0; r < 8; ++r) ind[base + r] = besti[r];
  }
}

// ---------- quantize gather + histogram + segment-sum (L2-resident) --------
__global__ void vq_scatter(const float* __restrict__ x, const float* __restrict__ embed,
                           const int* __restrict__ ind, float* __restrict__ quant,
                           float* __restrict__ cnt, float* __restrict__ embed_sum)
{
  const int row = blockIdx.x;
  const int d   = threadIdx.x;
  const int idx = ind[row];
  quant[(size_t)row * D_DIM + d] = embed[(size_t)idx * D_DIM + d];
  atomicAdd(&embed_sum[(size_t)idx * D_DIM + d], x[(size_t)row * D_DIM + d]);
  if (d == 0) atomicAdd(&cnt[idx], 1.0f);
}

// ---------------- EMA cluster size + smoothing + perplexity ----------------
__global__ void vq_finalize_cs(const float* __restrict__ cluster_size,
                               const float* __restrict__ cnt,
                               float* __restrict__ out_ncs, float* __restrict__ out_perp,
                               float* __restrict__ smoothed)
{
  __shared__ float red[K_CODES];
  __shared__ float nsh;
  const int t = threadIdx.x;
  const float c   = cnt[t];
  const float ncs = cluster_size[t] * DECAY_F + OMD_F * c;
  out_ncs[t] = ncs;
  red[t] = ncs;
  __syncthreads();
  for (int s = K_CODES / 2; s > 0; s >>= 1) { if (t < s) red[t] += red[t + s]; __syncthreads(); }
  if (t == 0) nsh = red[0];
  __syncthreads();
  const float nsum = nsh;
  smoothed[t] = (ncs + EPS_F) / (nsum + (float)K_CODES * EPS_F) * nsum;
  const float prob = c / (float)N_ROWS;
  red[t] = prob * logf(prob + 1e-10f);
  __syncthreads();
  for (int s = K_CODES / 2; s > 0; s >>= 1) { if (t < s) red[t] += red[t + s]; __syncthreads(); }
  if (t == 0) out_perp[0] = expf(-red[0]);
}

// ------------------------- EMA embed avg + new embed -----------------------
__global__ void vq_finalize_embed(const float* __restrict__ embed_avg,
                                  const float* __restrict__ embed_sum,
                                  const float* __restrict__ smoothed,
                                  float* __restrict__ out_nea, float* __restrict__ out_ne)
{
  const int k = blockIdx.x;
  const int d = threadIdx.x;
  const size_t i = (size_t)k * D_DIM + d;
  const float nea = embed_avg[i] * DECAY_F + OMD_F * embed_sum[i];
  out_nea[i] = nea;
  out_ne[i]  = nea / smoothed[k];
}

// ---------------------------------------------------------------------------
extern "C" void kernel_launch(void* const* d_in, const int* in_sizes, int n_in,
                              void* d_out, int out_size, void* d_ws, size_t ws_size,
                              hipStream_t stream)
{
  const float* x            = (const float*)d_in[0];   // [16,4096,256]
  const float* embed        = (const float*)d_in[1];   // [1024,256]
  const float* cluster_size = (const float*)d_in[2];   // [1024]
  const float* embed_avg    = (const float*)d_in[3];   // [1024,256]

  // outputs concatenated flat in return order
  float* out       = (float*)d_out;
  float* quant     = out;                                        // 16777216
  float* perp      = out + (size_t)16777216;                     // 1
  float* new_embed = out + (size_t)16777217;                     // 262144
  float* new_cs    = out + (size_t)16777217 + 262144;            // 1024
  float* new_ea    = new_cs + 1024;                              // 262144

  // workspace layout
  int*   ind      = (int*)d_ws;                                  // 65536 ints
  float* fws      = (float*)((char*)d_ws + (size_t)N_ROWS * sizeof(int));
  float* norms05  = fws;                                         // 1024
  float* cnt      = fws + 1024;                                  // 1024
  float* embed_sum= cnt + 1024;                                  // 262144 (contiguous with cnt)
  float* smoothed = embed_sum + (size_t)K_CODES * D_DIM;         // 1024

  const int nzero = 1024 + K_CODES * D_DIM;                      // cnt + embed_sum
  vq_zero<<<(nzero + 255) / 256, 256, 0, stream>>>(cnt, nzero);
  vq_norms<<<(K_CODES + 255) / 256, 256, 0, stream>>>(embed, norms05);
  vq_argmax<<<N_ROWS / 128, 256, 0, stream>>>(x, embed, norms05, ind);
  vq_scatter<<<N_ROWS, D_DIM, 0, stream>>>(x, embed, ind, quant, cnt, embed_sum);
  vq_finalize_cs<<<1, K_CODES, 0, stream>>>(cluster_size, cnt, new_cs, perp, smoothed);
  vq_finalize_embed<<<K_CODES, D_DIM, 0, stream>>>(embed_avg, embed_sum, smoothed, new_ea, new_embed);
}
